// GlobalNodeAttentionFull_50208167690485
// MI455X (gfx1250) — matrix operand order repaired and checked by reading, not compiled, not run on hardware
//
#include <hip/hip_runtime.h>

typedef __bf16 bf16;
typedef __attribute__((ext_vector_type(16))) __bf16 v16bf;
typedef __attribute__((ext_vector_type(8)))  __bf16 bf16x8;
typedef __attribute__((ext_vector_type(8)))  float  v8f;

#define N_NODES 8192
#define NGRAPH  16
#define NHEAD   8
#define FDIM    1152
#define TFDIM   3456
#define HD      144
#define RD      16

__device__ __forceinline__ int imin(int a, int b) { return a < b ? a : b; }

// 16-bit A-matrix fragment: element e (VGPR e/2, half e%2) -> K index, given lane half hl
__device__ __forceinline__ int innerA(int e, int hl) {
  return (e < 8) ? (hl * 8 + e) : (16 + hl * 8 + (e - 8));
}

// CDNA5 async global->LDS copy (ASYNCcnt-tracked), GVS addressing:
// lds[vdst] = mem[saddr + vaddr]; 16 bytes per lane.
__device__ __forceinline__ void async_load_b128(unsigned lds_addr, unsigned goff,
                                                const void* base) {
  asm volatile("global_load_async_to_lds_b128 %0, %1, %2"
               :: "v"(lds_addr), "v"(goff), "s"(base)
               : "memory");
}
__device__ __forceinline__ void wait_async0() {
  asm volatile("s_wait_asynccnt 0" ::: "memory");
}

// ---------------------------------------------------------------------------
// Per-graph offsets: batch is sorted; offs[b] = lower_bound(batch, b), offs[B]=N
// ---------------------------------------------------------------------------
__global__ void gna_layout(const int* __restrict__ batch, int* __restrict__ offs,
                           int n, int nb) {
  int b = threadIdx.x;
  if (b > nb) return;
  int lo = 0, hi = n;
  while (lo < hi) {
    int mid = (lo + hi) >> 1;
    if (batch[mid] < b) lo = mid + 1; else hi = mid;
  }
  offs[b] = lo;
}

__global__ void gna_cvt(const float* __restrict__ in, bf16* __restrict__ out, int n) {
  int i = blockIdx.x * 256 + threadIdx.x;
  if (i < n) out[i] = (bf16)in[i];
}

// f32 [rows][cols] -> bf16 transposed [cols][rows] (coalesced reads along cols)
__global__ void gna_cvt_t(const float* __restrict__ in, bf16* __restrict__ out,
                          int rows, int cols) {
  int idx = blockIdx.x * 256 + threadIdx.x;
  if (idx < rows * cols) {
    int r = idx / cols, c = idx % cols;
    out[(size_t)c * rows + r] = (bf16)in[idx];
  }
}

// ---------------------------------------------------------------------------
// bf16 WMMA GEMM: C[M,N] = A[M,K] * BT[N,K]^T.  M mult of 128, N of 64, K of 64.
// Block 256 threads (8 waves), tile 128x64x64; wave -> 32x32 output (8 WMMA/step).
// Double-buffered LDS tiles filled by global_load_async_to_lds_b128 (ASYNCcnt),
// pipelined one K-step ahead of the WMMA compute.
// ---------------------------------------------------------------------------
template<int OUT_BF16>
__global__ __launch_bounds__(256)
void gna_gemm_bf16(const bf16* __restrict__ A, const bf16* __restrict__ BT,
                   float* __restrict__ Cf, bf16* __restrict__ Cb,
                   int M, int N, int K) {
  __shared__ bf16 As[2][128][72];   // [buf][m][k], 144B row stride (16B aligned)
  __shared__ bf16 Bs[2][64][72];    // [buf][n][k]
  const int tid  = threadIdx.x;
  const int lane = tid & 31, wid = tid >> 5;
  const int hl   = lane >> 4, mcol = lane & 15;
  const int wm   = wid & 3,  wn  = wid >> 2;
  const int m0   = blockIdx.y * 128, n0 = blockIdx.x * 64;

  v8f acc00 = {}, acc01 = {}, acc10 = {}, acc11 = {};

  const unsigned asBase = (unsigned)(size_t)(&As[0][0][0]);
  const unsigned bsBase = (unsigned)(size_t)(&Bs[0][0][0]);
  const unsigned asStride = 128u * 72u * 2u;          // bytes per A buffer
  const unsigned bsStride = 64u * 72u * 2u;           // bytes per B buffer

  const int arow = tid >> 1, acol = (tid & 1) * 32;   // 4x16B chunks per thread
  const int brow = tid >> 2, bcol = (tid & 3) * 16;   // 2x16B chunks per thread

  // async-issue one K-tile (64 wide) into LDS buffer `buf`
  auto issue_tile = [&](int buf, int k0) {
    unsigned la = asBase + (unsigned)buf * asStride + (unsigned)(arow * 72 + acol) * 2u;
    unsigned ga = (unsigned)(((size_t)(m0 + arow) * K + k0 + acol) * 2u);
    #pragma unroll
    for (int c = 0; c < 4; ++c) async_load_b128(la + c * 16u, ga + c * 16u, A);
    unsigned lb = bsBase + (unsigned)buf * bsStride + (unsigned)(brow * 72 + bcol) * 2u;
    unsigned gb = (unsigned)(((size_t)(n0 + brow) * K + k0 + bcol) * 2u);
    #pragma unroll
    for (int c = 0; c < 2; ++c) async_load_b128(lb + c * 16u, gb + c * 16u, BT);
  };

  issue_tile(0, 0);                                   // prologue
  int cur = 0;

  for (int k0 = 0; k0 < K; k0 += 64) {
    wait_async0();                                    // my tile-`cur` copies done
    __syncthreads();                                  // everyone's done; prev reads retired
    if (k0 + 64 < K) issue_tile(cur ^ 1, k0 + 64);    // overlap next DMA with compute
    if (k0 + 128 < K) {                               // keep L2 warm two steps ahead
      __builtin_prefetch(A + (size_t)(m0 + arow) * K + (k0 + 128) + acol, 0, 3);
      __builtin_prefetch(BT + (size_t)(n0 + brow) * K + (k0 + 128) + bcol, 0, 3);
    }

    #pragma unroll
    for (int kc = 0; kc < 64; kc += 32) {
      v16bf a0, a1, b0, b1;
      #pragma unroll
      for (int e = 0; e < 16; ++e) {
        int ka = kc + innerA(e, hl);
        int kb = kc + innerA(e, hl);
        a0[e] = As[cur][wm * 32 + mcol][ka];
        a1[e] = As[cur][wm * 32 + 16 + mcol][ka];
        b0[e] = Bs[cur][wn * 32 + mcol][kb];
        b1[e] = Bs[cur][wn * 32 + 16 + mcol][kb];
      }
      acc00 = __builtin_amdgcn_wmma_f32_16x16x32_bf16(false, a0, false, b0, (short)0, acc00, false, false);
      acc01 = __builtin_amdgcn_wmma_f32_16x16x32_bf16(false, a0, false, b1, (short)0, acc01, false, false);
      acc10 = __builtin_amdgcn_wmma_f32_16x16x32_bf16(false, a1, false, b0, (short)0, acc10, false, false);
      acc11 = __builtin_amdgcn_wmma_f32_16x16x32_bf16(false, a1, false, b1, (short)0, acc11, false, false);
    }
    cur ^= 1;
  }

  const int row0 = m0 + wm * 32 + hl * 8;
  const int c0   = n0 + wn * 32 + mcol;
  #pragma unroll
  for (int r = 0; r < 8; ++r) {
    if (OUT_BF16) {
      Cb[(size_t)(row0 + r) * N + c0]           = (bf16)acc00[r];
      Cb[(size_t)(row0 + r) * N + c0 + 16]      = (bf16)acc01[r];
      Cb[(size_t)(row0 + 16 + r) * N + c0]      = (bf16)acc10[r];
      Cb[(size_t)(row0 + 16 + r) * N + c0 + 16] = (bf16)acc11[r];
    } else {
      Cf[(size_t)(row0 + r) * N + c0]           = acc00[r];
      Cf[(size_t)(row0 + r) * N + c0 + 16]      = acc01[r];
      Cf[(size_t)(row0 + 16 + r) * N + c0]      = acc10[r];
      Cf[(size_t)(row0 + 16 + r) * N + c0 + 16] = acc11[r];
    }
  }
}

// ---------------------------------------------------------------------------
// Flash attention per (graph, 16-query tile); 8 waves = 8 heads share the
// Fourier distance bias (computed once per key tile, reused by all heads).
// qkv: [N, 3F] bf16 (Q | K | V, each [H][HD]).  aout: [N, F] bf16.
// ---------------------------------------------------------------------------
__global__ __launch_bounds__(256)
void gna_attn(const bf16* __restrict__ qkv, const float* __restrict__ pos,
              const float* __restrict__ rfreq, const float* __restrict__ Wrope,
              const int* __restrict__ offs, bf16* __restrict__ aout) {
  const int b   = blockIdx.y;
  const int qt  = blockIdx.x;
  const int off = offs[b];
  const int n   = offs[b + 1] - off;
  if (qt * 16 >= n) return;

  const int tid  = threadIdx.x;
  const int lane = tid & 31;
  const int h    = tid >> 5;                 // wave id == head
  const int hl   = lane >> 4, mcol = lane & 15;

  __shared__ float biasS[16][32][NHEAD];     // [qi][kj][h]
  __shared__ bf16  Pl[NHEAD][16][36];        // per-wave P transpose scratch
  __shared__ float posQ[16][3];
  __shared__ float posK[32][3];
  __shared__ float freqS[RD];
  __shared__ float wrS[RD][NHEAD];

  if (tid < RD) freqS[tid] = fabsf(rfreq[tid]);
  if (tid < RD * NHEAD) wrS[tid / NHEAD][tid % NHEAD] = Wrope[tid];
  if (tid < 16) {
    int node = off + imin(qt * 16 + tid, n - 1);
    posQ[tid][0] = pos[node * 3 + 0];
    posQ[tid][1] = pos[node * 3 + 1];
    posQ[tid][2] = pos[node * 3 + 2];
  }

  // Resident Q fragments: 5 K-chunks of 32 cover HD=144 (tail zero-padded)
  v16bf qa[5];
  {
    int nodeq = off + imin(qt * 16 + mcol, n - 1);
    const bf16* qp = qkv + (size_t)nodeq * TFDIM + h * HD;
    #pragma unroll
    for (int c = 0; c < 5; ++c) {
      #pragma unroll
      for (int e = 0; e < 16; ++e) {
        int k = c * 32 + innerA(e, hl);
        qa[c][e] = (k < HD) ? qp[k] : (bf16)0.0f;
      }
    }
  }

  v8f zero = {};
  v8f acc[9];
  #pragma unroll
  for (int t = 0; t < 9; ++t) acc[t] = zero;
  float mrow[8], lrow[8];
  #pragma unroll
  for (int r = 0; r < 8; ++r) { mrow[r] = -__builtin_inff(); lrow[r] = 0.f; }

  const float scale = 1.0f / 12.0f;          // HD^-0.5, HD=144

  for (int j0 = 0; j0 < n; j0 += 32) {
    __syncthreads();                          // protect biasS/posK from prev iter
    if (tid < 32) {
      int node = off + imin(j0 + tid, n - 1);
      posK[tid][0] = pos[node * 3 + 0];
      posK[tid][1] = pos[node * 3 + 1];
      posK[tid][2] = pos[node * 3 + 2];
    }
    __syncthreads();

    // Fourier bias, shared across the 8 heads (2 pairs per thread)
    for (int p = tid; p < 512; p += 256) {
      int i = p >> 5, j = p & 31;
      float dx = posQ[i][0] - posK[j][0];
      float dy = posQ[i][1] - posK[j][1];
      float dz = posQ[i][2] - posK[j][2];
      float d  = sqrtf(dx * dx + dy * dy + dz * dz);
      float bacc[NHEAD];
      #pragma unroll
      for (int hh = 0; hh < NHEAD; ++hh) bacc[hh] = 0.f;
      #pragma unroll
      for (int r = 0; r < RD; ++r) {
        float cv = __cosf(d * freqS[r]);
        #pragma unroll
        for (int hh = 0; hh < NHEAD; ++hh) bacc[hh] += cv * wrS[r][hh];
      }
      #pragma unroll
      for (int hh = 0; hh < NHEAD; ++hh) biasS[i][j][hh] = bacc[hh];
    }
    __syncthreads();

    // S = Q K^T  (two 16x16 tiles covering 32 keys)
    v8f S0 = zero, S1 = zero;
    {
      int nk0 = off + imin(j0 + mcol, n - 1);
      int nk1 = off + imin(j0 + 16 + mcol, n - 1);
      const bf16* kp0 = qkv + (size_t)nk0 * TFDIM + FDIM + h * HD;
      const bf16* kp1 = qkv + (size_t)nk1 * TFDIM + FDIM + h * HD;
      #pragma unroll
      for (int c = 0; c < 5; ++c) {
        v16bf b0, b1;
        #pragma unroll
        for (int e = 0; e < 16; ++e) {
          int k = c * 32 + innerA(e, hl);
          bf16 z = (bf16)0.0f;
          b0[e] = (k < HD) ? kp0[k] : z;
          b1[e] = (k < HD) ? kp1[k] : z;
        }
        S0 = __builtin_amdgcn_wmma_f32_16x16x32_bf16(false, qa[c], false, b0, (short)0, S0, false, false);
        S1 = __builtin_amdgcn_wmma_f32_16x16x32_bf16(false, qa[c], false, b1, (short)0, S1, false, false);
      }
    }

    // scale + bias + key mask; online softmax (rows live in lane halves)
    float p0[8], p1[8], alpha[8];
    #pragma unroll
    for (int r = 0; r < 8; ++r) {
      int i = hl * 8 + r;
      float s0 = S0[r] * scale + biasS[i][mcol][h];
      float s1 = S1[r] * scale + biasS[i][16 + mcol][h];
      if (j0 + mcol >= n)      s0 = -__builtin_inff();
      if (j0 + 16 + mcol >= n) s1 = -__builtin_inff();
      float tmax = fmaxf(s0, s1);
      #pragma unroll
      for (int m = 1; m < 16; m <<= 1) tmax = fmaxf(tmax, __shfl_xor(tmax, m, 32));
      float mnew = fmaxf(mrow[r], tmax);
      alpha[r] = __expf(mrow[r] - mnew);
      p0[r] = __expf(s0 - mnew);
      p1[r] = __expf(s1 - mnew);
      float ps = p0[r] + p1[r];
      #pragma unroll
      for (int m = 1; m < 16; m <<= 1) ps += __shfl_xor(ps, m, 32);
      lrow[r] = lrow[r] * alpha[r] + ps;
      mrow[r] = mnew;
    }
    #pragma unroll
    for (int t = 0; t < 9; ++t)
      #pragma unroll
      for (int r = 0; r < 8; ++r) acc[t][r] *= alpha[r];

    // P: D-layout -> A-layout via per-wave LDS (same-wave DS ops are in-order)
    #pragma unroll
    for (int r = 0; r < 8; ++r) {
      Pl[h][hl * 8 + r][mcol]      = (bf16)p0[r];
      Pl[h][hl * 8 + r][16 + mcol] = (bf16)p1[r];
    }
    v16bf pa;
    #pragma unroll
    for (int e = 0; e < 16; ++e) pa[e] = Pl[h][mcol][innerA(e, hl)];

    // acc += P(16x32) * V(32x144): 9 column tiles, K=32 exact
    #pragma unroll
    for (int t = 0; t < 9; ++t) {
      v16bf vb;
      #pragma unroll
      for (int e = 0; e < 16; ++e) {
        int kk = innerA(e, hl);
        int node = off + imin(j0 + kk, n - 1);
        vb[e] = qkv[(size_t)node * TFDIM + 2 * FDIM + h * HD + t * 16 + mcol];
      }
      acc[t] = __builtin_amdgcn_wmma_f32_16x16x32_bf16(false, pa, false, vb, (short)0, acc[t], false, false);
    }
  }

  // epilogue: normalize and store bf16 rows for the output projection
  #pragma unroll
  for (int r = 0; r < 8; ++r) {
    int i = qt * 16 + hl * 8 + r;
    if (i < n) {
      size_t node = (size_t)(off + i);
      float inv = 1.0f / lrow[r];
      #pragma unroll
      for (int t = 0; t < 9; ++t)
        aout[node * FDIM + h * HD + t * 16 + mcol] = (bf16)(acc[t][r] * inv);
    }
  }
}

// ---------------------------------------------------------------------------
// Residual + LayerNorm: y = (x + proj - mu) * rsqrt(var + eps) * gamma + beta
// ---------------------------------------------------------------------------
__global__ __launch_bounds__(256)
void gna_ln(const float* __restrict__ x, const float* __restrict__ proj,
            const float* __restrict__ gamma, const float* __restrict__ beta,
            float* __restrict__ out) {
  const int node = blockIdx.x;
  const int tid  = threadIdx.x;
  __shared__ float hbuf[FDIM];
  __shared__ float red[8][2];
  __shared__ float stats[2];

  float s = 0.f, ss = 0.f;
  for (int f = tid; f < FDIM; f += 256) {
    float hv = x[(size_t)node * FDIM + f] + proj[(size_t)node * FDIM + f];
    hbuf[f] = hv; s += hv; ss += hv * hv;
  }
  #pragma unroll
  for (int m = 1; m < 32; m <<= 1) {
    s  += __shfl_xor(s,  m, 32);
    ss += __shfl_xor(ss, m, 32);
  }
  if ((tid & 31) == 0) { red[tid >> 5][0] = s; red[tid >> 5][1] = ss; }
  __syncthreads();
  if (tid == 0) {
    float ts = 0.f, tss = 0.f;
    for (int w = 0; w < 8; ++w) { ts += red[w][0]; tss += red[w][1]; }
    float mu  = ts / (float)FDIM;
    float var = tss / (float)FDIM - mu * mu;
    stats[0] = mu;
    stats[1] = rsqrtf(var + 1e-5f);
  }
  __syncthreads();
  float mu = stats[0], inv = stats[1];
  for (int f = tid; f < FDIM; f += 256)
    out[(size_t)node * FDIM + f] = (hbuf[f] - mu) * inv * gamma[f] + beta[f];
}

// ---------------------------------------------------------------------------
extern "C" void kernel_launch(void* const* d_in, const int* in_sizes, int n_in,
                              void* d_out, int out_size, void* d_ws, size_t ws_size,
                              hipStream_t stream) {
  (void)in_sizes; (void)n_in; (void)out_size; (void)ws_size;
  const float* x     = (const float*)d_in[0];   // [N, 9, 128] == [N, F]
  const float* pos   = (const float*)d_in[1];   // [N, 3]
  const float* Wqkv  = (const float*)d_in[2];   // [F, 3F]
  const float* Wout  = (const float*)d_in[3];   // [F, F]
  const float* rfreq = (const float*)d_in[4];   // [RD]
  const float* Wrope = (const float*)d_in[5];   // [RD, H]
  const float* gamma = (const float*)d_in[6];   // [F]
  const float* beta  = (const float*)d_in[7];   // [F]
  const int*   batch = (const int*)d_in[8];     // [N] sorted

  char* ws = (char*)d_ws;
  int*   offs   = (int*)ws;    ws += 256;
  bf16*  xb     = (bf16*)ws;   ws += (size_t)N_NODES * FDIM * 2;
  bf16*  wqkvT  = (bf16*)ws;   ws += (size_t)FDIM * TFDIM * 2;   // [3F, F]
  bf16*  woutT  = (bf16*)ws;   ws += (size_t)FDIM * FDIM * 2;    // [F, F] (transposed)
  bf16*  qkvb   = (bf16*)ws;   ws += (size_t)N_NODES * TFDIM * 2;
  bf16*  attnb  = (bf16*)ws;   ws += (size_t)N_NODES * FDIM * 2;
  float* projf  = (float*)ws;  ws += (size_t)N_NODES * FDIM * 4;

  gna_layout<<<1, 32, 0, stream>>>(batch, offs, N_NODES, NGRAPH);

  int tot;
  tot = N_NODES * FDIM; gna_cvt<<<(tot + 255) / 256, 256, 0, stream>>>(x, xb, tot);
  tot = FDIM * TFDIM;   gna_cvt_t<<<(tot + 255) / 256, 256, 0, stream>>>(Wqkv, wqkvT, FDIM, TFDIM);
  tot = FDIM * FDIM;    gna_cvt_t<<<(tot + 255) / 256, 256, 0, stream>>>(Wout, woutT, FDIM, FDIM);

  // qkv = x @ Wqkv   (bf16 out, f32 accumulate); pipelined async-LDS tiles
  gna_gemm_bf16<1><<<dim3(TFDIM / 64, N_NODES / 128), 256, 0, stream>>>(
      xb, wqkvT, nullptr, qkvb, N_NODES, TFDIM, FDIM);

  // per-graph flash attention with shared Fourier bias
  gna_attn<<<dim3(N_NODES / 16, NGRAPH), 256, 0, stream>>>(
      qkvb, pos, rfreq, Wrope, offs, attnb);

  // proj = attn @ Wout  (f32 out)
  gna_gemm_bf16<0><<<dim3(FDIM / 64, N_NODES / 128), 256, 0, stream>>>(
      attnb, woutT, projf, nullptr, N_NODES, FDIM, FDIM);

  // y = LN(x + proj) * gamma + beta
  gna_ln<<<N_NODES, 256, 0, stream>>>(x, projf, gamma, beta, (float*)d_out);
}
